// PhraseCompressor_8615704396089
// MI455X (gfx1250) — matrix operand order repaired and checked by reading, not compiled, not run on hardware
//
#include <hip/hip_runtime.h>
#include <hip/hip_bf16.h>

// ---------------------------------------------------------------------------
// PhraseCompressor for MI455X (gfx1250, wave32, WMMA)
//   h:        (B=4, T=4096, D=2048) f32
//   mask:     (B, P=1024, L=8) bool (1 byte)
//   idx:      (B, P, L) i32
//   W_kv,W_z: (c=128, D) f32
//   B_pos:    (L, c) f32
//   out:      (B, P, c) f32
// ---------------------------------------------------------------------------

typedef __attribute__((ext_vector_type(16))) __bf16 v16bf;
typedef __attribute__((ext_vector_type(8)))  __bf16 v8bf;
typedef __attribute__((ext_vector_type(8)))  float  v8f;

#define Dv      2048
#define Tv      4096
#define Pv      1024
#define Cv      128
#define Lv      8
#define KT      64          // Dv / 32
#define SA      2056        // padded LDS row stride (bf16 elems): kills bank conflicts
#define NEG_INF (-1e30f)

static __device__ __forceinline__ v16bf cat16(v8bf lo, v8bf hi) {
  return __builtin_shufflevector(lo, hi, 0,1,2,3,4,5,6,7,8,9,10,11,12,13,14,15);
}

static __device__ __forceinline__ v8f wmma_bf16(v16bf a, v16bf b, v8f c) {
  // D = A(16x32) * B(32x16) + C, f32 accumulate
  return __builtin_amdgcn_wmma_f32_16x16x32_bf16(false, a, false, b, (short)0, c,
                                                 false, false);
}

// ---------------------------------------------------------------------------
// Kernel 0: pack W_kv||W_z into bf16 WMMA B-fragments.
// Fragment (w, kt): 16x32 tile, channels w*16..w*16+15, k = kt*32..kt*32+31.
// Lane layout (B, 32x16, contiguous k order, same as A in pc_main): lanes 0-15 hold K=kt*32+0..15
// (N = lane), lanes 16-31 hold K=kt*32+16..31 (N = lane-16); 16 bf16 per lane.
// 1 KB per fragment, fragments contiguous: offset = (w*KT + kt)*512 elems.
// ---------------------------------------------------------------------------
__global__ void pc_pack_weights(const float* __restrict__ Wkv,
                                const float* __restrict__ Wz,
                                __bf16* __restrict__ pack) {
  const int frag = blockIdx.x;          // 0..16*KT-1
  const int w    = frag >> 6;
  const int kt   = frag & (KT - 1);
  const int lane = threadIdx.x;         // 0..31
  const int cc   = w * 16 + (lane & 15);
  const float* Wrow = (cc < Cv) ? (Wkv + (size_t)cc * Dv)
                                : (Wz  + (size_t)(cc - Cv) * Dv);
  const int k0 = kt * 32 + ((lane & 16) ? 16 : 0);
  __bf16* dst = pack + (size_t)frag * 512 + lane * 16;
#pragma unroll
  for (int e = 0; e < 16; ++e) dst[e] = (__bf16)Wrow[k0 + e];
}

// ---------------------------------------------------------------------------
// Main kernel: 512 threads (16 waves). Each block = 4 phrases = 32 rows.
// ---------------------------------------------------------------------------
__global__ __launch_bounds__(512)
void pc_main(const float* __restrict__ h,
             const unsigned char* __restrict__ mask,
             const int* __restrict__ pidx,
             const float* __restrict__ Bpos,
             const __bf16* __restrict__ pack,
             float* __restrict__ out) {
  extern __shared__ char smem[];
  __bf16* A   = (__bf16*)smem;      // [32][SA] gathered bf16 token rows
  float*  ACC = (float*)smem;       // aliased after K-loop: [32][256] f32

  const int g   = blockIdx.x;       // 0..1023
  const int q0  = g * 4;            // global phrase base (b*P + p)
  const int tid = threadIdx.x;

  // ---- Stage 1: gather 32 token rows, convert f32 -> bf16 into LDS ----
  for (int i = tid; i < 32 * (Dv / 4); i += 512) {
    const int row  = i >> 9;                 // Dv/4 == 512
    const int col4 = i & 511;
    const int q = q0 + (row >> 3);
    const int l = row & 7;
    const int b = q >> 10;                   // P == 1024
    const int t = pidx[q * Lv + l];
    const float4 v =
        *(const float4*)(h + ((size_t)b * Tv + t) * Dv + (size_t)col4 * 4);
    __bf16* dst = A + (size_t)row * SA + col4 * 4;
    dst[0] = (__bf16)v.x; dst[1] = (__bf16)v.y;
    dst[2] = (__bf16)v.z; dst[3] = (__bf16)v.w;
  }
  __syncthreads();

  // ---- Stage 2: WMMA. Wave w owns N-tile w (0-7: kv, 8-15: z). ----
  const int wave = tid >> 5;
  const int lane = tid & 31;
  const int arow = lane & 15;
  const int koff = (lane & 16) ? 16 : 0;    // contiguous k order (16h+s), matches packed B
  const __bf16* bbase = pack + (size_t)wave * KT * 512 + lane * 16;

  v8f acc0 = {};
  v8f acc1 = {};
  for (int kt = 0; kt < KT; ++kt) {
    // prefetch B two fragments ahead (global_prefetch path)
    const int ktp = (kt + 2 < KT) ? kt + 2 : KT - 1;
    __builtin_prefetch((const void*)(bbase + (size_t)ktp * 512), 0, 1);

    // B fragment: 32 contiguous bytes per lane
    const v8bf b0 = *(const v8bf*)(bbase + (size_t)kt * 512);
    const v8bf b1 = *(const v8bf*)(bbase + (size_t)kt * 512 + 8);
    const v16bf bb = cat16(b0, b1);

    const int k = kt * 32 + koff;
    // M-tile 0: rows 0..15
    {
      const __bf16* ap = A + (size_t)arow * SA + k;
      const v16bf aa = cat16(*(const v8bf*)ap, *(const v8bf*)(ap + 8));
      acc0 = wmma_bf16(aa, bb, acc0);
    }
    // M-tile 1: rows 16..31
    {
      const __bf16* ap = A + (size_t)(16 + arow) * SA + k;
      const v16bf aa = cat16(*(const v8bf*)ap, *(const v8bf*)(ap + 8));
      acc1 = wmma_bf16(aa, bb, acc1);
    }
  }
  __syncthreads();   // everyone done reading A before aliasing it with ACC

  // ---- Stage 3: spill accumulators to LDS [row 0..31][chan 0..255] ----
  // C/D layout: lanes 0-15: N=lane, VGPR v -> M=v; lanes 16-31: N=lane-16, M=v+8
  const int colbase =
      ((wave < 8) ? wave * 16 : Cv + (wave - 8) * 16) + (lane & 15);
  const int radd = (lane & 16) ? 8 : 0;
#pragma unroll
  for (int v = 0; v < 8; ++v) {
    ACC[(v + radd) * 256 + colbase]        = acc0[v];
    ACC[(16 + v + radd) * 256 + colbase]   = acc1[v];
  }
  __syncthreads();

  // ---- Stage 4: masked softmax over L=8 slots + weighted reduce ----
  const int j  = tid >> 7;            // phrase within block (0..3)
  const int cc = tid & (Cv - 1);      // channel
  const int q  = q0 + j;

  float z[Lv];
  float mx = NEG_INF;
  bool any = false;
#pragma unroll
  for (int l = 0; l < Lv; ++l) {
    const bool mm = mask[q * Lv + l] != 0;
    any = any || mm;
    const float zz = mm
        ? (ACC[(j * Lv + l) * 256 + Cv + cc] + Bpos[l * Cv + cc])
        : NEG_INF;
    z[l] = zz;
    mx = fmaxf(mx, zz);
  }
  float s = 0.f, num = 0.f;
#pragma unroll
  for (int l = 0; l < Lv; ++l) {
    const float e = __expf(z[l] - mx);   // unmasked -> exp(-huge) == 0 exactly
    s   += e;
    num += e * ACC[(j * Lv + l) * 256 + cc];
  }
  out[(size_t)q * Cv + cc] = any ? (num / s) : 0.f;
}

// ---------------------------------------------------------------------------
extern "C" void kernel_launch(void* const* d_in, const int* in_sizes, int n_in,
                              void* d_out, int out_size, void* d_ws, size_t ws_size,
                              hipStream_t stream) {
  const float*         h    = (const float*)d_in[0];
  const unsigned char* mask = (const unsigned char*)d_in[1]; // jax bool = 1 byte
  const int*           pidx = (const int*)d_in[2];
  const float*         Wkv  = (const float*)d_in[3];
  const float*         Wz   = (const float*)d_in[4];
  const float*         Bpos = (const float*)d_in[5];
  float*               out  = (float*)d_out;

  __bf16* pack = (__bf16*)d_ws;   // 256*2048 bf16 = 1 MiB packed B-fragments

  // Pack weights into WMMA-ready bf16 fragments (16 N-tiles x 64 K-tiles).
  pc_pack_weights<<<16 * KT, 32, 0, stream>>>(Wkv, Wz, pack);

  // Main fused gather + dual-projection + masked-softmax + reduce.
  const size_t lds_bytes = (size_t)32 * SA * sizeof(__bf16);  // 131584 B
  pc_main<<<(4 * Pv) / 4, 512, lds_bytes, stream>>>(h, mask, pidx, Bpos, pack, out);
}
